// TimeDelayLayer_v1_6356551598869
// MI455X (gfx1250) — matrix operand order repaired and checked by reading, not compiled, not run on hardware
//
#include <hip/hip_runtime.h>

#define B_DIM   64
#define C_INCH  512
#define T_IN    2048
#define C_OUT   512
#define KWID    5
#define T_OUT   (T_IN - KWID + 1)   // 2044
#define T_PAD   2080                // 65 * 32, covers t0max(2032)+15+kw(4)=2051
#define KSTEPS  (KWID * (C_INCH / 32))   // 80

typedef __attribute__((ext_vector_type(8)))  _Float16 v8h;
typedef __attribute__((ext_vector_type(16))) _Float16 v16h;
typedef __attribute__((ext_vector_type(8)))  float    v8f;

// ---------------------------------------------------------------------------
// Pre-pass 1: x [B][C_in][T] f32 -> xt [B][T_PAD][C_in] f16 (zero pad t>=T_IN)
// LDS-tiled 32x32 transpose: coalesced reads over t, coalesced writes over ci.
// ---------------------------------------------------------------------------
__global__ void __launch_bounds__(256)
tdl_transpose_x(const float* __restrict__ x, _Float16* __restrict__ xt) {
  __shared__ _Float16 tile[32][33];          // +1 pad: conflict-free transpose

  const int tx = threadIdx.x & 31;
  const int ty = threadIdx.x >> 5;           // 0..7
  const int t0  = blockIdx.x * 32;           // 65 tiles -> t in [0,2080)
  const int ci0 = blockIdx.y * 32;           // 16 tiles
  const int b   = blockIdx.z;

  // Read: lanes sweep t (contiguous f32), rows sweep ci.
#pragma unroll
  for (int r = 0; r < 4; ++r) {
    int ci = ci0 + ty + r * 8;
    int t  = t0 + tx;
    float v = 0.0f;
    if (t < T_IN)
      v = __builtin_nontemporal_load(&x[((size_t)(b * C_INCH + ci) * T_IN) + t]);
    tile[ty + r * 8][tx] = (_Float16)v;
  }
  __syncthreads();

  // Write: lanes sweep ci (contiguous f16), rows sweep t.  t0+31 < 2080 always.
#pragma unroll
  for (int r = 0; r < 4; ++r) {
    int t  = t0 + ty + r * 8;
    int ci = ci0 + tx;
    xt[((size_t)b * T_PAD + t) * C_INCH + ci] = tile[tx][ty + r * 8];
  }
}

// ---------------------------------------------------------------------------
// Pre-pass 2: w [C_out][C_in][KW] f32 -> whs: WMMA-A fragment-ready layout.
// whs[mt][ks][lane][e] (16 halves per lane): one v16h load per fragment,
// fully coalesced (1KB contiguous per wave), sequential over the K loop.
//   mt = co tile (co = mt*16 + lane%16), ks = kw*16 + ci0/32,
//   A 16x32 f16, contiguous k order chosen here: K = 16*(lane/16)+e
// ---------------------------------------------------------------------------
__global__ void tdl_swizzle_w(const float* __restrict__ w,
                              _Float16* __restrict__ whs) {
  int i = blockIdx.x * blockDim.x + threadIdx.x;
  if (i >= (C_OUT / 16) * KSTEPS * 32 * 16) return;   // 1,310,720
  int e    = i & 15;
  int r    = i >> 4;
  int lane = r & 31;
  int r2   = r >> 5;
  int ks   = r2 % KSTEPS;
  int mt   = r2 / KSTEPS;

  int kw  = ks >> 4;
  int ci0 = (ks & 15) << 5;
  int lh  = lane >> 4;
  int co  = mt * 16 + (lane & 15);
  int kl  = 16 * lh + e;
  int ci  = ci0 + kl;
  whs[i] = (_Float16)w[((size_t)(co * C_INCH + ci)) * KWID + kw];
}

// ---------------------------------------------------------------------------
// Main implicit-GEMM conv kernel.
// Block: 256 threads = 8 waves (2 M-waves x 4 N-waves), tile 128(M) x 128(N).
// Wave tile: 64(M) x 32(N) = 4x2 accumulator fragments.
// K loop: 80 steps of 32 (kw-major, ci inner), 8 WMMA per step.
// ---------------------------------------------------------------------------
__global__ void __launch_bounds__(256)
tdl_wmma_gemm(const _Float16* __restrict__ xt,
              const _Float16* __restrict__ whs,
              const float* __restrict__ bias,
              float* __restrict__ out) {
  const int lane = threadIdx.x & 31;
  const int wave = threadIdx.x >> 5;
  const int wm   = wave & 1;           // M-wave 0..1
  const int wn   = wave >> 1;          // N-wave 0..3
  const int l16  = lane & 15;
  const int lh   = lane >> 4;

  const int m_base  = blockIdx.x * 128 + wm * 64;
  const int mt_base = blockIdx.x * 8 + wm * 4;       // 16-row tile index
  const int nt0     = blockIdx.y * 8 + wn * 2;       // 16-col N tile index

  // B fragment pointers: 32x16 f16, same contiguous k order as A: lane -> N col = l16,
  // K = 16*lh + e -> one contiguous 32B run per lane.
  int bb[2], tt0[2];
  const _Float16* pB[2];
#pragma unroll
  for (int ni = 0; ni < 2; ++ni) {
    int nt  = nt0 + ni;
    bb[ni]  = nt >> 7;                 // 128 t-tiles per batch
    tt0[ni] = (nt & 127) << 4;
    pB[ni]  = xt + ((size_t)(bb[ni] * T_PAD + tt0[ni] + l16) * C_INCH) + lh * 16;
  }

  // A fragment pointers: fragment-ready stream, 512 halves (1KB) per K-step.
  const _Float16* pA[4];
#pragma unroll
  for (int mi = 0; mi < 4; ++mi)
    pA[mi] = whs + ((size_t)(mt_base + mi) * KSTEPS * 512) + lane * 16;

  v8f acc[4][2] = {};

  for (int kw = 0; kw < KWID; ++kw) {
    const int koff = kw * C_INCH;      // t-shift by kw == +kw*C_INCH in xt rows
#pragma unroll 2
    for (int cb = 0; cb < 16; ++cb) {
      const int o  = koff + (cb << 5);         // B offset (elements)
      const int ks = (kw << 4) + cb;           // A K-step index

      v16h bf0 = *(const v16h*)(pB[0] + o);
      v16h bf1 = *(const v16h*)(pB[1] + o);

      v16h af[4];
#pragma unroll
      for (int mi = 0; mi < 4; ++mi)
        af[mi] = *(const v16h*)(pA[mi] + (size_t)ks * 512);

#pragma unroll
      for (int mi = 0; mi < 4; ++mi) {
        acc[mi][0] = __builtin_amdgcn_wmma_f32_16x16x32_f16(
            false, af[mi], false, bf0, (short)0, acc[mi][0], false, false);
        acc[mi][1] = __builtin_amdgcn_wmma_f32_16x16x32_f16(
            false, af[mi], false, bf1, (short)0, acc[mi][1], false, false);
      }
    }
  }

  // Epilogue: D 16x16 f32 layout: lane -> N col = l16, VGPR r -> M = r + 8*lh.
#pragma unroll
  for (int ni = 0; ni < 2; ++ni) {
    int t = tt0[ni] + l16;
    if (t < T_OUT) {
#pragma unroll
      for (int mi = 0; mi < 4; ++mi) {
#pragma unroll
        for (int r = 0; r < 8; ++r) {
          int   co = m_base + mi * 16 + r + 8 * lh;
          float v  = acc[mi][ni][r] + bias[co];
          __builtin_nontemporal_store(
              v, &out[((size_t)(bb[ni] * C_OUT + co)) * T_OUT + t]);
        }
      }
    }
  }
}

// ---------------------------------------------------------------------------
extern "C" void kernel_launch(void* const* d_in, const int* in_sizes, int n_in,
                              void* d_out, int out_size, void* d_ws, size_t ws_size,
                              hipStream_t stream) {
  const float* x    = (const float*)d_in[0];
  const float* kern = (const float*)d_in[1];
  const float* bias = (const float*)d_in[2];
  float*       out  = (float*)d_out;

  // Workspace: [0, 2.62MB) whs ; [4MB, 4MB+136.3MB) xt
  _Float16* whs = (_Float16*)d_ws;
  _Float16* xt  = (_Float16*)((char*)d_ws + ((size_t)4 << 20));

  {
    int total = (C_OUT / 16) * KSTEPS * 32 * 16;          // 1,310,720
    tdl_swizzle_w<<<(total + 255) / 256, 256, 0, stream>>>(kern, whs);
  }
  {
    dim3 grid(T_PAD / 32, C_INCH / 32, B_DIM);            // (65, 16, 64)
    tdl_transpose_x<<<grid, 256, 0, stream>>>(x, xt);
  }
  {
    dim3 grid(C_OUT / 128, (B_DIM * (T_IN / 16)) / 8);    // (4, 1024)
    tdl_wmma_gemm<<<grid, 256, 0, stream>>>(xt, whs, bias, out);
  }
}
